// MultiHeadAttention_82918638616729
// MI455X (gfx1250) — matrix operand order repaired and checked by reading, not compiled, not run on hardware
//
#include <hip/hip_runtime.h>
#include <hip/hip_bf16.h>
#include <stdint.h>

typedef __attribute__((ext_vector_type(16))) __bf16 v16bf;
typedef __attribute__((ext_vector_type(8)))  float  v8f;

#define S_LEN 2048
#define NHEAD 16
#define HDIM  64
#define EMB   1024

// ---------------------------------------------------------------------------
// WMMA wrapper: D = A(16x32 bf16) x B(32x16 bf16) + C(16x16 f32)
// ---------------------------------------------------------------------------
__device__ __forceinline__ v8f wmma_bf16(v16bf a, v16bf b, v8f c) {
  return __builtin_amdgcn_wmma_f32_16x16x32_bf16(false, a, false, b,
                                                 (short)0, c, false, false);
}

// ---------------------------------------------------------------------------
// CDNA5 async copy: 16 bytes global -> LDS per lane, tracked by ASYNCcnt.
// (GLOBAL_LOAD_ASYNC_TO_LDS_B128, ISA 15.18.3 op 98)
// ---------------------------------------------------------------------------
__device__ __forceinline__ void async_copy16(uint32_t lds_addr, const void* gaddr) {
  asm volatile("global_load_async_to_lds_b128 %0, %1, off"
               :: "v"(lds_addr), "v"((uint64_t)(uintptr_t)gaddr) : "memory");
}
__device__ __forceinline__ void wait_async() {
  asm volatile("s_wait_asynccnt 0x0" ::: "memory");
}

// ---------------------------------------------------------------------------
// Fragment loaders (wave32; contiguous k = 16*(lane/16) + s in both A and B)
// ---------------------------------------------------------------------------
__device__ __forceinline__ v16bf load_a_f32(const float* __restrict__ A, int lda,
                                            int row, int k0, int l16) {
  v16bf a;
  const int kbase = l16 * 16;
  const float* base = A + (size_t)row * lda + k0;
#pragma unroll
  for (int v = 0; v < 8; ++v) {
    int kOff = kbase + 2 * v;
    a[2 * v]     = (__bf16)base[kOff];
    a[2 * v + 1] = (__bf16)base[kOff + 1];
  }
  return a;
}

__device__ __forceinline__ v16bf load_a_bf16(const __bf16* __restrict__ A, int lda,
                                             int row, int k0, int l16) {
  v16bf a;
  const int kbase = l16 * 16;
  const __bf16* base = A + (size_t)row * lda + k0;
#pragma unroll
  for (int v = 0; v < 8; ++v) {
    int kOff = kbase + 2 * v;
    a[2 * v]     = base[kOff];
    a[2 * v + 1] = base[kOff + 1];
  }
  return a;
}

// B fragment where this lane's 16 K-values are contiguous bf16 in memory
// (lane col = lane&15, K-half = (lane>>4)*16): vectorizes to b128 loads.
__device__ __forceinline__ v16bf load_b16_contig(const __bf16* p) {
  v16bf b;
#pragma unroll
  for (int v = 0; v < 16; ++v) b[v] = p[v];
  return b;
}

// B fragment with K-major stride (used for V tile in LDS): B[k][n].
__device__ __forceinline__ v16bf load_b_strided(const __bf16* base, int stride,
                                                int ln, int l16) {
  v16bf b;
#pragma unroll
  for (int v = 0; v < 8; ++v) {
    int k = l16 * 16 + 2 * v;
    b[2 * v]     = base[(size_t)k * stride + ln];
    b[2 * v + 1] = base[(size_t)(k + 1) * stride + ln];
  }
  return b;
}

// A fragment from LDS-staged P (row-major 16x32 bf16)
__device__ __forceinline__ v16bf load_a_lds(const __bf16* P, int ln, int l16) {
  v16bf a;
  const int kbase = l16 * 16;
#pragma unroll
  for (int v = 0; v < 8; ++v) {
    int kOff = kbase + 2 * v;
    a[2 * v]     = P[ln * 32 + kOff];
    a[2 * v + 1] = P[ln * 32 + kOff + 1];
  }
  return a;
}

// ---------------------------------------------------------------------------
// Kernel 0: weight transpose + bf16 convert:  WT[n][k] = (bf16) W[k][n]
// ---------------------------------------------------------------------------
__global__ __launch_bounds__(256) void wtrans_kernel(const float* __restrict__ W,
                                                     __bf16* __restrict__ WT,
                                                     int Kdim, int N) {
  int idx = blockIdx.x * 256 + threadIdx.x;
  if (idx >= Kdim * N) return;
  int r = idx / N, c = idx - r * N;
  WT[(size_t)c * Kdim + r] = (__bf16)W[idx];
}

// ---------------------------------------------------------------------------
// Kernel 1: QKV projection.  Wave computes a 64x32 tile (8 WMMA / K-step).
// C = X(f32) @ W + b, stored bf16 head-de-interleaved:
//   col c -> (h=c&15, d=c>>4);  out[((b*16+h)*S + s)*64 + d]
// ---------------------------------------------------------------------------
__global__ __launch_bounds__(128) void proj_kernel(
    const float* __restrict__ X, const __bf16* __restrict__ WT,
    const float* __restrict__ bias, __bf16* __restrict__ out,
    int M, int Kdim, int N) {
  const int lane = threadIdx.x & 31;
  const int ln = lane & 15, l16 = lane >> 4;
  const int wid = blockIdx.x * 4 + (threadIdx.x >> 5);
  const int nT = N >> 5;                       // 32-wide column tiles
  const int mTile = wid / nT, nTile = wid % nT;
  const int m0 = mTile * 64, n0 = nTile * 32;
  if (m0 >= M) return;

  v8f acc[4][2] = {};
  for (int k0 = 0; k0 < Kdim; k0 += 32) {
    v16bf a[4], b[2];
#pragma unroll
    for (int mi = 0; mi < 4; ++mi)
      a[mi] = load_a_f32(X, Kdim, m0 + 16 * mi + ln, k0, l16);
#pragma unroll
    for (int ni = 0; ni < 2; ++ni)
      b[ni] = load_b16_contig(WT + (size_t)(n0 + 16 * ni + ln) * Kdim + k0 + l16 * 16);
#pragma unroll
    for (int mi = 0; mi < 4; ++mi)
#pragma unroll
      for (int ni = 0; ni < 2; ++ni)
        acc[mi][ni] = wmma_bf16(a[mi], b[ni], acc[mi][ni]);
  }

  const int bb = m0 >> 11;                     // m0 / S_LEN (tile never crosses)
  const int s0 = m0 & (S_LEN - 1);
#pragma unroll
  for (int ni = 0; ni < 2; ++ni) {
    const int c = n0 + 16 * ni + ln;
    const int h = c & 15, d = c >> 4;
    const float bc = bias[c];
    __bf16* op = out + (((size_t)bb * NHEAD + h) * S_LEN + s0 + l16 * 8) * HDIM + d;
#pragma unroll
    for (int mi = 0; mi < 4; ++mi)
#pragma unroll
      for (int r = 0; r < 8; ++r)
        op[(size_t)(16 * mi + r) * HDIM] = (__bf16)(acc[mi][ni][r] + bc);
  }
}

// ---------------------------------------------------------------------------
// Kernel 2: flash attention per (b,h); block = 4 waves = 64 query rows.
// K/V tiles staged to LDS with global_load_async_to_lds_b128.
// ---------------------------------------------------------------------------
__global__ __launch_bounds__(128) void attn_kernel(
    const __bf16* __restrict__ qbuf, const __bf16* __restrict__ kbuf,
    const __bf16* __restrict__ vbuf, const int* __restrict__ mask,
    __bf16* __restrict__ obuf) {
  const int tid = threadIdx.x;
  const int lane = tid & 31;
  const int ln = lane & 15, l16 = lane >> 4;
  const int w = tid >> 5;
  const int qBlocks = S_LEN / 64;
  const int bh = blockIdx.x / qBlocks;
  const int q0 = (blockIdx.x % qBlocks) * 64 + w * 16;
  const int b = bh >> 4, h = bh & 15;

  const __bf16* Q  = qbuf + (size_t)bh * S_LEN * HDIM;
  const __bf16* Km = kbuf + (size_t)bh * S_LEN * HDIM;
  const __bf16* V  = vbuf + (size_t)bh * S_LEN * HDIM;
  const int* Mb    = mask + (size_t)b * S_LEN * S_LEN;

  __shared__ __bf16 ktile[32 * HDIM];          // 4 KB
  __shared__ __bf16 vtile[32 * HDIM];          // 4 KB
  __shared__ __bf16 psh[4][16 * 32];           // 4 KB

  const v16bf qa0 = load_a_bf16(Q, HDIM, q0 + ln, 0, l16);
  const v16bf qa1 = load_a_bf16(Q, HDIM, q0 + ln, 32, l16);

  float mprev[8], lsum[8];
  v8f acc[4] = {};
#pragma unroll
  for (int r = 0; r < 8; ++r) { mprev[r] = -INFINITY; lsum[r] = 0.f; }

  const float scale = 0.125f;                  // 1/sqrt(64)

  for (int j0 = 0; j0 < S_LEN; j0 += 32) {
    // --- stage K(32x64) + V(32x64) tiles: 512 x 16B chunks / 128 threads ---
#pragma unroll
    for (int i = 0; i < 4; ++i) {
      int c = tid + 128 * i;                   // 0..511
      int row = c >> 3, sub = (c & 7) * 8;     // 8 bf16 = 16B per chunk
      const __bf16* src = ((row < 32) ? Km : (V - 32 * HDIM))
                          + (size_t)(j0 + row) * HDIM + sub;
      __bf16* dst = ((row < 32) ? ktile : (vtile - 32 * HDIM)) + row * HDIM + sub;
      async_copy16((uint32_t)(uintptr_t)dst, src);
    }
    wait_async();
    __syncthreads();

    // --- energy = Q K^T for 32 kv columns (two 16-wide halves) ---
    v16bf kb00 = load_b16_contig(ktile + (ln)      * HDIM + 0  + l16 * 16);
    v16bf kb01 = load_b16_contig(ktile + (ln)      * HDIM + 32 + l16 * 16);
    v16bf kb10 = load_b16_contig(ktile + (16 + ln) * HDIM + 0  + l16 * 16);
    v16bf kb11 = load_b16_contig(ktile + (16 + ln) * HDIM + 32 + l16 * 16);
    v8f e0 = {}, e1 = {};
    e0 = wmma_bf16(qa0, kb00, e0);
    e0 = wmma_bf16(qa1, kb01, e0);
    e1 = wmma_bf16(qa0, kb10, e1);
    e1 = wmma_bf16(qa1, kb11, e1);

    // --- mask (before scale, like reference) + online softmax ---
#pragma unroll
    for (int r = 0; r < 8; ++r) {
      int mrow = q0 + r + l16 * 8;
      const int* mp = Mb + (size_t)mrow * S_LEN + j0;
      if (mp[ln] == 0)      e0[r] = -1e20f;
      if (mp[16 + ln] == 0) e1[r] = -1e20f;
      e0[r] *= scale;
      e1[r] *= scale;
      float t = fmaxf(e0[r], e1[r]);
      t = fmaxf(t, __shfl_xor(t, 1));
      t = fmaxf(t, __shfl_xor(t, 2));
      t = fmaxf(t, __shfl_xor(t, 4));
      t = fmaxf(t, __shfl_xor(t, 8));          // row max within 16-lane half
      float mnew = fmaxf(mprev[r], t);
      float p0 = __expf(e0[r] - mnew);
      float p1 = __expf(e1[r] - mnew);
      float ps = p0 + p1;
      ps += __shfl_xor(ps, 1);
      ps += __shfl_xor(ps, 2);
      ps += __shfl_xor(ps, 4);
      ps += __shfl_xor(ps, 8);                 // row sum
      float alpha = __expf(mprev[r] - mnew);
      lsum[r] = lsum[r] * alpha + ps;
      mprev[r] = mnew;
#pragma unroll
      for (int cI = 0; cI < 4; ++cI) acc[cI][r] *= alpha;
      int mloc = r + l16 * 8;
      psh[w][mloc * 32 + ln]      = (__bf16)p0;
      psh[w][mloc * 32 + 16 + ln] = (__bf16)p1;
    }
    __syncthreads();

    // --- out += P x V ---
    v16bf pa  = load_a_lds(&psh[w][0], ln, l16);
    v16bf vb0 = load_b_strided(vtile + 0,  HDIM, ln, l16);
    v16bf vb1 = load_b_strided(vtile + 16, HDIM, ln, l16);
    v16bf vb2 = load_b_strided(vtile + 32, HDIM, ln, l16);
    v16bf vb3 = load_b_strided(vtile + 48, HDIM, ln, l16);
    acc[0] = wmma_bf16(pa, vb0, acc[0]);
    acc[1] = wmma_bf16(pa, vb1, acc[1]);
    acc[2] = wmma_bf16(pa, vb2, acc[2]);
    acc[3] = wmma_bf16(pa, vb3, acc[3]);
    __syncthreads();                           // tiles reused next iteration
  }

  // --- normalize, store concat-order [b, s, h*64 + d] as bf16 ---
  float inv[8];
#pragma unroll
  for (int r = 0; r < 8; ++r) inv[r] = 1.0f / lsum[r];
  __bf16* ob = obuf + (size_t)(b * S_LEN + q0 + l16 * 8) * (NHEAD * HDIM) + h * HDIM;
#pragma unroll
  for (int cI = 0; cI < 4; ++cI)
#pragma unroll
    for (int r = 0; r < 8; ++r)
      ob[(size_t)r * (NHEAD * HDIM) + cI * 16 + ln] = (__bf16)(acc[cI][r] * inv[r]);
}

// ---------------------------------------------------------------------------
// Kernel 3: output projection.  Wave computes 64x32 tile; f32 result + bias.
// ---------------------------------------------------------------------------
__global__ __launch_bounds__(128) void oproj_kernel(
    const __bf16* __restrict__ A, const __bf16* __restrict__ WT,
    const float* __restrict__ bias, float* __restrict__ out,
    int M, int Kdim, int N) {
  const int lane = threadIdx.x & 31;
  const int ln = lane & 15, l16 = lane >> 4;
  const int wid = blockIdx.x * 4 + (threadIdx.x >> 5);
  const int nT = N >> 5;
  const int mTile = wid / nT, nTile = wid % nT;
  const int m0 = mTile * 64, n0 = nTile * 32;
  if (m0 >= M) return;

  v8f acc[4][2] = {};
  for (int k0 = 0; k0 < Kdim; k0 += 32) {
    v16bf a[4], b[2];
#pragma unroll
    for (int mi = 0; mi < 4; ++mi)
      a[mi] = load_a_bf16(A, Kdim, m0 + 16 * mi + ln, k0, l16);
#pragma unroll
    for (int ni = 0; ni < 2; ++ni)
      b[ni] = load_b16_contig(WT + (size_t)(n0 + 16 * ni + ln) * Kdim + k0 + l16 * 16);
#pragma unroll
    for (int mi = 0; mi < 4; ++mi)
#pragma unroll
      for (int ni = 0; ni < 2; ++ni)
        acc[mi][ni] = wmma_bf16(a[mi], b[ni], acc[mi][ni]);
  }
#pragma unroll
  for (int ni = 0; ni < 2; ++ni) {
    const int c = n0 + 16 * ni + ln;
    const float bc = bias[c];
    float* op = out + (size_t)(m0 + l16 * 8) * N + c;
#pragma unroll
    for (int mi = 0; mi < 4; ++mi)
#pragma unroll
      for (int r = 0; r < 8; ++r)
        op[(size_t)(16 * mi + r) * N] = acc[mi][ni][r] + bc;
  }
}

// ---------------------------------------------------------------------------
extern "C" void kernel_launch(void* const* d_in, const int* in_sizes, int n_in,
                              void* d_out, int out_size, void* d_ws, size_t ws_size,
                              hipStream_t stream) {
  const float* queries = (const float*)d_in[0];
  const float* keys    = (const float*)d_in[1];
  const float* values  = (const float*)d_in[2];
  const int*   mask    = (const int*)d_in[3];
  const float* Wq = (const float*)d_in[4];
  const float* bq = (const float*)d_in[5];
  const float* Wk = (const float*)d_in[6];
  const float* bk = (const float*)d_in[7];
  const float* Wv = (const float*)d_in[8];
  const float* bv = (const float*)d_in[9];
  const float* Wo = (const float*)d_in[10];
  const float* bo = (const float*)d_in[11];
  float* out = (float*)d_out;

  const int B = 2;
  const int M = B * S_LEN;                                  // 4096
  const size_t headElems = (size_t)B * NHEAD * S_LEN * HDIM;  // 4M bf16

  __bf16* qbuf = (__bf16*)d_ws;
  __bf16* kbuf = qbuf + headElems;
  __bf16* vbuf = kbuf + headElems;
  __bf16* obuf = vbuf + headElems;                          // [B,S,H*D]
  __bf16* wtq  = obuf + (size_t)M * EMB;
  __bf16* wtk  = wtq + (size_t)EMB * EMB;
  __bf16* wtv  = wtk + (size_t)EMB * EMB;
  __bf16* wto  = wtv + (size_t)EMB * EMB;                   // total ~42 MB

  dim3 blk(128);
  dim3 grdT((EMB * EMB + 255) / 256);
  wtrans_kernel<<<grdT, dim3(256), 0, stream>>>(Wq, wtq, EMB, EMB);
  wtrans_kernel<<<grdT, dim3(256), 0, stream>>>(Wk, wtk, EMB, EMB);
  wtrans_kernel<<<grdT, dim3(256), 0, stream>>>(Wv, wtv, EMB, EMB);
  wtrans_kernel<<<grdT, dim3(256), 0, stream>>>(Wo, wto, EMB, EMB);

  dim3 grdGemm(((M / 64) * (EMB / 32)) / 4);                // 512 blocks
  proj_kernel<<<grdGemm, blk, 0, stream>>>(queries, wtq, bq, qbuf, M, EMB, NHEAD * HDIM);
  proj_kernel<<<grdGemm, blk, 0, stream>>>(keys,    wtk, bk, kbuf, M, EMB, NHEAD * HDIM);
  proj_kernel<<<grdGemm, blk, 0, stream>>>(values,  wtv, bv, vbuf, M, EMB, NHEAD * HDIM);

  dim3 grdAttn(B * NHEAD * (S_LEN / 64));                   // 1024 blocks
  attn_kernel<<<grdAttn, blk, 0, stream>>>(qbuf, kbuf, vbuf, mask, obuf);

  oproj_kernel<<<grdGemm, blk, 0, stream>>>(obuf, wto, bo, out, M, NHEAD * HDIM, EMB);
}
